// MLAttention_50835232915801
// MI455X (gfx1250) — matrix operand order repaired and checked by reading, not compiled, not run on hardware
//
#include <hip/hip_runtime.h>

// ---------------------------------------------------------------------------
// Problem constants (match reference setup_inputs; top_k hardcoded to 128).
// ---------------------------------------------------------------------------
#define S_LEN 2048
#define D_MODEL 1024
#define Q_IN 512
#define HI 16
#define DI 64
#define NH 8
#define DQK 64
#define DV 64
#define TOPK 128
#define NEG_INF (-1e30f)
#define LN_EPS 1e-5f

typedef __attribute__((ext_vector_type(16))) __bf16 v16bf;
typedef __attribute__((ext_vector_type(8)))  __bf16 v8bf;
typedef __attribute__((ext_vector_type(8)))  float  v8f;

// ---------------------------------------------------------------------------
// Kernel 0: per-row indexer prep.
//   ki = LN(x @ w_k) * g + b   -> bf16 [S, DI]   (stored row-major = col-major
//                                                 for the scores GEMM B operand)
//   w  = (x @ w_weights) * HI^-0.5 -> f32 [S, HI]
// ---------------------------------------------------------------------------
__global__ __launch_bounds__(64) void indexer_prep(
    const float* __restrict__ x, const float* __restrict__ w_k,
    const float* __restrict__ ln_g, const float* __restrict__ ln_b,
    const float* __restrict__ w_weights,
    __bf16* __restrict__ ki_bf, float* __restrict__ wrow) {
  const int row = blockIdx.x;
  const int d = threadIdx.x;
  const float* xr = x + row * D_MODEL;

  float acc = 0.f;
  for (int c = 0; c < D_MODEL; ++c) acc = fmaf(xr[c], w_k[c * DI + d], acc);

  __shared__ float sh[DI];
  sh[d] = acc;
  __syncthreads();
  float mu = 0.f;
  #pragma unroll
  for (int i = 0; i < DI; ++i) mu += sh[i];
  mu *= (1.f / DI);
  float var = 0.f;
  #pragma unroll
  for (int i = 0; i < DI; ++i) { float t = sh[i] - mu; var = fmaf(t, t, var); }
  var *= (1.f / DI);
  float kin = (acc - mu) * rsqrtf(var + LN_EPS) * ln_g[d] + ln_b[d];
  ki_bf[row * DI + d] = (__bf16)kin;

  if (d < HI) {
    float wa = 0.f;
    for (int c = 0; c < D_MODEL; ++c) wa = fmaf(xr[c], w_weights[c * HI + d], wa);
    wrow[row * HI + d] = wa * 0.25f;  // HI^-0.5 = 16^-0.5
  }
}

// ---------------------------------------------------------------------------
// Kernel 1a: bulk f32 -> bf16 convert (A operand stays row-major).
// ---------------------------------------------------------------------------
__global__ void f32_to_bf16(const float* __restrict__ src,
                            __bf16* __restrict__ dst, int n) {
  int i = blockIdx.x * blockDim.x + threadIdx.x;
  if (i < n) dst[i] = (__bf16)src[i];
}

// ---------------------------------------------------------------------------
// Kernel 1b: transposing f32 -> bf16 convert: dst[c*rows + r] = src[r*cols + c].
// 32x32 LDS tile (33-stride to dodge bank conflicts), 256 threads = 32x8.
// Makes w_q column-major so the GEMM's B loads are contiguous b128s.
// ---------------------------------------------------------------------------
__global__ __launch_bounds__(256) void transpose_f32_to_bf16(
    const float* __restrict__ src, __bf16* __restrict__ dst,
    int rows, int cols) {
  __shared__ float t[32][33];
  const int tc = blockIdx.x, tr = blockIdx.y;
  const int lx = threadIdx.x & 31, ly = threadIdx.x >> 5;
  #pragma unroll
  for (int i = 0; i < 32; i += 8) {
    t[ly + i][lx] = src[(tr * 32 + ly + i) * cols + tc * 32 + lx];
  }
  __syncthreads();
  #pragma unroll
  for (int i = 0; i < 32; i += 8) {
    dst[(tc * 32 + ly + i) * rows + tr * 32 + lx] = (__bf16)t[lx][ly + i];
  }
}

// ---------------------------------------------------------------------------
// Kernel 2: bf16 WMMA GEMM, f32 accumulate, B stored COLUMN-major.
//   C[M,N] = A[M,K] @ B[K,N],  Bc(k,n) at Bc[n*ldb + k]
// Each wave owns a 16x32 output supertile (two 16x16 WMMA accumulators sharing
// the A operand): 2 independent v_wmma per K-step, all operand loads are
// contiguous 16B/32B vector loads.
// CAUSAL: supertiles fully above the diagonal skip the K-loop (wave-uniform)
// and store NEG_INF; diagonal tiles mask per element in the epilogue.
// VGPR layouts used here (wave32; A/B use contiguous K, not ISA table order):
//   A (bf16 16x32): lane L row M=L&15; elem e -> K = 16*(L>>4) + e
//   B (bf16 32x16): lane L col N=L&15; elem e -> K = 16*(L>>4) + e
//   C/D (f32 16x16): lane L col N=L&15; vgpr r -> row M = r + 8*(L>>4)
// ---------------------------------------------------------------------------
template <bool CAUSAL>
__global__ __launch_bounds__(256) void wmma_gemm_bf16_nt(
    const __bf16* __restrict__ A, const __bf16* __restrict__ Bc,
    float* __restrict__ C, int M, int N, int K,
    int lda, int ldb, int ldc) {
  const int lane = threadIdx.x & 31;
  const int wave = threadIdx.x >> 5;
  const int tiles_n = N >> 5;                      // 32-wide supertiles
  const int tile = blockIdx.x * 8 + wave;
  if (tile >= (M >> 4) * tiles_n) return;          // wave-uniform
  const int tm = tile / tiles_n;
  const int tn = tile - tm * tiles_n;
  const int n0 = tn << 5;

  const int mrow = (tm << 4) + (lane & 15);        // A row this lane supplies
  const int nco0 = n0 + (lane & 15);               // C col, tile 0
  const int nco1 = nco0 + 16;                      // C col, tile 1
  const int kbA = (lane >> 4) * 16;
  const int kbB = (lane >> 4) * 16;

  v8f c0 = {}, c1 = {};
  const bool all_masked = CAUSAL && (n0 > (tm << 4) + 15);
  if (!all_masked) {
    const __bf16* ap  = A  + mrow * lda + kbA;
    const __bf16* bp0 = Bc + nco0 * ldb + kbB;
    const __bf16* bp1 = Bc + nco1 * ldb + kbB;
    for (int k0 = 0; k0 < K; k0 += 32) {
      v8bf alo = *(const v8bf*)(ap + k0);
      v8bf ahi = *(const v8bf*)(ap + k0 + 8);
      v16bf a;
      #pragma unroll
      for (int e = 0; e < 8; ++e) { a[e] = alo[e]; a[8 + e] = ahi[e]; }
      v16bf b0 = *(const v16bf*)(bp0 + k0);
      v16bf b1 = *(const v16bf*)(bp1 + k0);
      c0 = __builtin_amdgcn_wmma_f32_16x16x32_bf16(false, a, false, b0,
                                                   (short)0, c0, false, false);
      c1 = __builtin_amdgcn_wmma_f32_16x16x32_bf16(false, a, false, b1,
                                                   (short)0, c1, false, false);
    }
  }

  #pragma unroll
  for (int r = 0; r < 8; ++r) {
    const int row = (tm << 4) + r + ((lane >> 4) << 3);
    float v0 = c0[r], v1 = c1[r];
    if (CAUSAL) {
      if (nco0 > row) v0 = NEG_INF;
      if (nco1 > row) v1 = NEG_INF;
    }
    C[row * ldc + nco0] = v0;
    C[row * ldc + nco1] = v1;
  }
}

// ---------------------------------------------------------------------------
// Kernel 3: q_eff[q,d] = DI^-0.5 * sum_h w[q,h] * qi[q, h*DI + d]  -> bf16
// ---------------------------------------------------------------------------
__global__ void combine_qeff(const float* __restrict__ qi,
                             const float* __restrict__ wrow,
                             __bf16* __restrict__ qeff) {
  int i = blockIdx.x * blockDim.x + threadIdx.x;
  if (i >= S_LEN * DI) return;
  int qpos = i >> 6, d = i & 63;
  float acc = 0.f;
  #pragma unroll
  for (int h = 0; h < HI; ++h)
    acc = fmaf(wrow[qpos * HI + h], qi[qpos * D_MODEL + h * DI + d], acc);
  qeff[i] = (__bf16)(acc * 0.125f);  // DI^-0.5
}

// ---------------------------------------------------------------------------
// Kernel 4: causal top-128 per row, iterative max-extraction in LDS.
// Lowest-index tie-break (matches lax.top_k). Extracted slots go below
// NEG_INF so masked entries are still selected when <128 valid keys exist;
// those get re-masked (zero weight) in the attention kernel.
// ---------------------------------------------------------------------------
__global__ __launch_bounds__(256) void topk_kernel(
    const float* __restrict__ scores, int* __restrict__ idx) {
  const int row = blockIdx.x;
  const int tid = threadIdx.x;
  __shared__ float s[S_LEN];
  __shared__ float rv[256];
  __shared__ int   ri[256];
  __builtin_prefetch(scores + row * S_LEN, 0, 1);
  for (int j = tid; j < S_LEN; j += 256) s[j] = scores[row * S_LEN + j];
  __syncthreads();
  for (int it = 0; it < TOPK; ++it) {
    float bv = -3e38f; int bi = 0x7fffffff;
    #pragma unroll
    for (int u = 0; u < 8; ++u) {
      int j = tid + u * 256;
      float vv = s[j];
      if (vv > bv || (vv == bv && j < bi)) { bv = vv; bi = j; }
    }
    rv[tid] = bv; ri[tid] = bi;
    __syncthreads();
    for (int stp = 128; stp > 0; stp >>= 1) {
      if (tid < stp) {
        float v2 = rv[tid + stp]; int i2 = ri[tid + stp];
        if (v2 > rv[tid] || (v2 == rv[tid] && i2 < ri[tid])) {
          rv[tid] = v2; ri[tid] = i2;
        }
      }
      __syncthreads();
    }
    if (tid == 0) { idx[row * TOPK + it] = ri[0]; s[ri[0]] = -3e38f; }
    __syncthreads();
  }
}

// ---------------------------------------------------------------------------
// Kernel 5: gathered sparse attention. Block = 128 threads = one (head,query).
// Thread j owns selected key j for QK^T; threads 0..63 own output dims for PV
// (V reads coalesced across dims). f32 throughout (matches reference).
// ---------------------------------------------------------------------------
__global__ __launch_bounds__(128) void sparse_attn(
    const float* __restrict__ q, const float* __restrict__ k,
    const float* __restrict__ v, const int* __restrict__ idx,
    float* __restrict__ out) {
  const int qpos = blockIdx.x;
  const int h = blockIdx.y;
  const int tid = threadIdx.x;
  __shared__ float qv[DQK];
  __shared__ float att[TOPK];
  __shared__ int   ji[TOPK];
  __shared__ float red[128];

  if (tid < DQK) qv[tid] = q[(qpos * NH + h) * DQK + tid];
  const int j = idx[qpos * TOPK + tid];
  ji[tid] = j;
  __syncthreads();

  float a;
  if (j <= qpos) {
    const float* kr = k + (j * NH + h) * DQK;
    float acc = 0.f;
    #pragma unroll
    for (int d = 0; d < DQK; ++d) acc = fmaf(qv[d], kr[d], acc);
    a = acc * 0.125f;  // DQK^-0.5
  } else {
    a = NEG_INF;
  }

  red[tid] = a; __syncthreads();
  for (int stp = 64; stp > 0; stp >>= 1) {
    if (tid < stp) red[tid] = fmaxf(red[tid], red[tid + stp]);
    __syncthreads();
  }
  const float m = red[0];
  __syncthreads();
  const float p = __expf(a - m);
  red[tid] = p; __syncthreads();
  for (int stp = 64; stp > 0; stp >>= 1) {
    if (tid < stp) red[tid] += red[tid + stp];
    __syncthreads();
  }
  const float inv = 1.f / red[0];
  __syncthreads();
  att[tid] = p * inv;
  __syncthreads();

  if (tid < DV) {
    float acc = 0.f;
    for (int jj = 0; jj < TOPK; ++jj)
      acc = fmaf(att[jj], v[(ji[jj] * NH + h) * DV + tid], acc);
    out[(qpos * NH + h) * DV + tid] = acc;
  }
}

// ---------------------------------------------------------------------------
// Host-side orchestration. Workspace layout (bytes, ~30 MB total):
//   [0,        8M)   qi          f32 [S, HI*DI]
//   [8M,      24M)   scores      f32 [S, S] (causal-masked)
//   [24M,     26M)   q_input     bf16 row-major [S, Q_IN]
//   [26M,     27M)   w_q^T       bf16 col-major [D_MODEL, Q_IN]
//   [27M, +256K)     ki          bf16 [S, DI]
//   [+256K]          q_eff       bf16 [S, DI]
//   [+128K]          wrow        f32 [S, HI]
//   [+1M]            idx         i32 [S, TOPK]
// ---------------------------------------------------------------------------
extern "C" void kernel_launch(void* const* d_in, const int* in_sizes, int n_in,
                              void* d_out, int out_size, void* d_ws, size_t ws_size,
                              hipStream_t stream) {
  const float* x         = (const float*)d_in[0];
  const float* q_input   = (const float*)d_in[1];
  const float* q         = (const float*)d_in[2];
  const float* k         = (const float*)d_in[3];
  const float* v         = (const float*)d_in[4];
  const float* w_q       = (const float*)d_in[5];
  const float* w_k       = (const float*)d_in[6];
  const float* ln_g      = (const float*)d_in[7];
  const float* ln_b      = (const float*)d_in[8];
  const float* w_weights = (const float*)d_in[9];
  float* out = (float*)d_out;

  char* ws = (char*)d_ws;
  const size_t MB = 1024u * 1024u;
  float*  qi      = (float*)(ws);
  float*  scores  = (float*)(ws + 8 * MB);
  __bf16* qin_bf  = (__bf16*)(ws + 24 * MB);
  __bf16* wqT_bf  = (__bf16*)(ws + 26 * MB);
  __bf16* ki_bf   = (__bf16*)(ws + 27 * MB);
  __bf16* qeff_bf = (__bf16*)(ws + 27 * MB + 256 * 1024);
  float*  wrow    = (float*)(ws + 27 * MB + 512 * 1024);
  int*    idx     = (int*)(ws + 27 * MB + 640 * 1024);

  // 1) ki layernorm + per-head weights
  indexer_prep<<<S_LEN, 64, 0, stream>>>(x, w_k, ln_g, ln_b, w_weights, ki_bf, wrow);

  // 2) bf16 operands: q_input row-major, w_q transposed to column-major
  f32_to_bf16<<<(S_LEN * Q_IN + 255) / 256, 256, 0, stream>>>(
      q_input, qin_bf, S_LEN * Q_IN);
  transpose_f32_to_bf16<<<dim3(D_MODEL / 32, Q_IN / 32), 256, 0, stream>>>(
      w_q, wqT_bf, Q_IN, D_MODEL);

  // 3) qi = q_input @ w_q   (2048x1024x512, bf16 WMMA, 16x32 per wave)
  {
    int tiles = (S_LEN / 16) * (D_MODEL / 32);
    wmma_gemm_bf16_nt<false><<<(tiles + 7) / 8, 256, 0, stream>>>(
        qin_bf, wqT_bf, qi, S_LEN, D_MODEL, Q_IN, Q_IN, Q_IN, D_MODEL);
  }

  // 4) q_eff = head-weighted sum of qi, scaled, bf16
  combine_qeff<<<(S_LEN * DI + 255) / 256, 256, 0, stream>>>(qi, wrow, qeff_bf);

  // 5) scores = q_eff @ ki^T with causal mask (2048x2048x64, bf16 WMMA)
  {
    int tiles = (S_LEN / 16) * (S_LEN / 32);
    wmma_gemm_bf16_nt<true><<<(tiles + 7) / 8, 256, 0, stream>>>(
        qeff_bf, ki_bf, scores, S_LEN, S_LEN, DI, DI, DI, S_LEN);
  }

  // 6) causal top-128 per query
  topk_kernel<<<S_LEN, 256, 0, stream>>>(scores, idx);

  // 7) gathered sparse attention -> out [1, S, H, DV]
  sparse_attn<<<dim3(S_LEN, NH), 128, 0, stream>>>(q, k, v, idx, out);
}
